// JanossyPoolingImproper_55198919688258
// MI455X (gfx1250) — matrix operand order repaired and checked by reading, not compiled, not run on hardware
//
#include <hip/hip_runtime.h>

#define N_IMP_C 300000

typedef __attribute__((ext_vector_type(16))) __bf16 v16bf;
typedef __attribute__((ext_vector_type(8)))  __bf16 v8bf;
typedef __attribute__((ext_vector_type(8)))  float  v8f;
typedef __attribute__((ext_vector_type(4)))  float  v4f;

// ---------------------------------------------------------------------------
// Workspace layout (bf16 elements in d_ws):
//   [0      , 16384) : W1a tiles  = W1[0:128]+W1[256:384]+W1[384:512]  (K=128,N=128)
//   [16384  , 32768) : W1b tiles  = 3*W1[128:256]                      (K=128,N=128)
//   [32768  , 49152) : W2 tiles                                        (K=128,N=128)
//   [49152  , 65536) : W3 tiles                                        (K=128,N=128)
//   [65536  , 67584) : Wo tiles padded to N=16 (cols >=6 are zero)     (K=128,N=16)
// Tile layout matches the CDNA5 WMMA B-matrix (32x16, 16-bit) lane layout:
//   element index e = ((kt*NT + nt)*32 + lane)*16 + j
//   lane<16 : holds column n = nt*16+lane, rows K = kt*32 + j (j<8), kt*32+16 + j-8 (j>=8)
//   lane>=16: holds column n = nt*16+lane-16, rows K = kt*32+8 + j (j<8), kt*32+24 + j-8 (j>=8)
// ---------------------------------------------------------------------------
__global__ void prep_weights(const float* __restrict__ W1,
                             const float* __restrict__ W2,
                             const float* __restrict__ W3,
                             const float* __restrict__ Wo,
                             __bf16* __restrict__ ws) {
  int e = blockIdx.x * blockDim.x + threadIdx.x;   // 0..16383
  if (e >= 16384) return;
  const int j  = e & 15;
  const int l  = (e >> 4) & 31;
  const int nt = (e >> 9) & 7;
  const int kt = e >> 12;
  const int k  = kt * 32 + ((l >> 4) << 3) + (j & 7) + ((j >> 3) << 4);    // K index 0..127
  const int n  = nt * 16 + (l & 15);               // N index 0..127

  float a = W1[k * 128 + n] + W1[(256 + k) * 128 + n] + W1[(384 + k) * 128 + n];
  ws[e]         = (__bf16)a;
  ws[16384 + e] = (__bf16)(3.0f * W1[(128 + k) * 128 + n]);
  ws[32768 + e] = (__bf16)W2[k * 128 + n];
  ws[49152 + e] = (__bf16)W3[k * 128 + n];
  if (nt == 0) {                                   // head weights, N padded 6->16
    int eo = (kt * 32 + l) * 16 + j;
    ws[65536 + eo] = (n < 6) ? (__bf16)Wo[k * 6 + n] : (__bf16)0.0f;
  }
}

// B fragment: 32 contiguous bytes per lane (pre-swizzled in ws).
__device__ __forceinline__ v16bf load_bfrag(const __bf16* __restrict__ wb,
                                            int tile, int lane) {
  return *(const v16bf*)(wb + (tile * 32 + lane) * 16);
}

// A fragments for 16 rows x K=128 from a per-wave 16x136 bf16 LDS tile.
// 16-bit A layout (ISA 7.12.2): lane<16 -> row=lane, elems 0..7 = K+0..7,
// elems 8..15 = K+16..23; lane>=16 -> row=lane-16, K offsets +8 / +24.
__device__ __forceinline__ void load_afrags(const __bf16* __restrict__ A,
                                            v16bf aF[4], int lane) {
  const int kh = lane >> 4;
  const int rl = lane & 15;
#pragma unroll
  for (int kf = 0; kf < 4; ++kf) {
    const __bf16* p = A + rl * 136 + kf * 32 + kh * 8;
    v8bf lo = *(const v8bf*)(p);
    v8bf hi = *(const v8bf*)(p + 16);
    aF[kf] = __builtin_shufflevector(lo, hi, 0, 1, 2, 3, 4, 5, 6, 7,
                                     8, 9, 10, 11, 12, 13, 14, 15);
  }
}

// C/D layout: VGPR j -> row j (lanes 0-15) / row j+8 (lanes 16-31), col = lane&15.
__device__ __forceinline__ void epilogue_relu_store(const v8f& acc, float bv,
                                                    __bf16* __restrict__ A,
                                                    int nt, int lane) {
  const int kh = lane >> 4;
  const int nl = lane & 15;
#pragma unroll
  for (int j = 0; j < 8; ++j) {
    float v = acc[j] + bv;
    v = v > 0.0f ? v : 0.0f;
    A[(j + kh * 8) * 136 + nt * 16 + nl] = (__bf16)v;
  }
}

// 4-step K chain with all B fragments preloaded before the WMMA chain, so
// only the first load's latency is exposed (the acc chain serializes anyway).
__device__ __forceinline__ v8f chain4(const v16bf aF[4],
                                      const __bf16* __restrict__ wb,
                                      int nt, int lane, v8f acc) {
  v16bf b0 = load_bfrag(wb, 0 * 8 + nt, lane);
  v16bf b1 = load_bfrag(wb, 1 * 8 + nt, lane);
  v16bf b2 = load_bfrag(wb, 2 * 8 + nt, lane);
  v16bf b3 = load_bfrag(wb, 3 * 8 + nt, lane);
  acc = __builtin_amdgcn_wmma_f32_16x16x32_bf16(false, aF[0], false, b0,
                                                (short)0, acc, false, false);
  acc = __builtin_amdgcn_wmma_f32_16x16x32_bf16(false, aF[1], false, b1,
                                                (short)0, acc, false, false);
  acc = __builtin_amdgcn_wmma_f32_16x16x32_bf16(false, aF[2], false, b2,
                                                (short)0, acc, false, false);
  acc = __builtin_amdgcn_wmma_f32_16x16x32_bf16(false, aF[3], false, b3,
                                                (short)0, acc, false, false);
  return acc;
}

__device__ __forceinline__ void mlp_layer(const v16bf aF[4],
                                          const __bf16* __restrict__ wb,
                                          const float* __restrict__ bias,
                                          __bf16* __restrict__ A, int lane) {
  const int nl = lane & 15;
#pragma unroll
  for (int nt = 0; nt < 8; ++nt) {
    v8f acc = {};
    acc = chain4(aF, wb, nt, lane, acc);
    epilogue_relu_store(acc, bias[nt * 16 + nl], A, nt, lane);
  }
}

__global__ __launch_bounds__(256, 1)
void janossy_mlp(const float* __restrict__ h,
                 const int* __restrict__ idx0, const int* __restrict__ idx1,
                 const int* __restrict__ idx2, const int* __restrict__ idx3,
                 const float* __restrict__ b1, const float* __restrict__ b2,
                 const float* __restrict__ b3, const float* __restrict__ bo,
                 const __bf16* __restrict__ wq, float* __restrict__ out) {
  // One 16x136 bf16 activation tile per wave (8 waves). No cross-wave deps,
  // DS in-order per wave makes a single buffer per layer round-trip safe.
  __shared__ __bf16 act[8][16 * 136];

  const int lane = threadIdx.x & 31;
  const int wv   = threadIdx.x >> 5;
  __bf16* A = &act[wv][0];
  const int kh = lane >> 4;
  const int rl = lane & 15;
  const int rowbase = blockIdx.x * 128 + wv * 16;

  // --- gather + fold: build layer-1 A fragments straight from global h ---
  int grow = rowbase + rl;
  int rs = grow < N_IMP_C ? grow : 0;           // branchless clamp, EXEC stays full
  const float* p0 = h + idx0[rs] * 128;
  const float* p1 = h + idx1[rs] * 128;
  const float* p2 = h + idx2[rs] * 128;
  const float* p3 = h + idx3[rs] * 128;

  v16bf aS[4], aT[4];                            // s = h0+h2+h3 ; t = h1
#pragma unroll
  for (int kf = 0; kf < 4; ++kf) {
    const int c0 = kf * 32 + kh * 8;
    float sv[16], tv[16];
#pragma unroll
    for (int ch = 0; ch < 2; ++ch) {
      const int c = c0 + ch * 16;
      v4f a0 = *(const v4f*)(p0 + c); v4f a0b = *(const v4f*)(p0 + c + 4);
      v4f a1 = *(const v4f*)(p1 + c); v4f a1b = *(const v4f*)(p1 + c + 4);
      v4f a2 = *(const v4f*)(p2 + c); v4f a2b = *(const v4f*)(p2 + c + 4);
      v4f a3 = *(const v4f*)(p3 + c); v4f a3b = *(const v4f*)(p3 + c + 4);
#pragma unroll
      for (int u = 0; u < 4; ++u) {
        sv[ch * 8 + u]     = a0[u]  + a2[u]  + a3[u];
        sv[ch * 8 + 4 + u] = a0b[u] + a2b[u] + a3b[u];
        tv[ch * 8 + u]     = a1[u];
        tv[ch * 8 + 4 + u] = a1b[u];
      }
    }
#pragma unroll
    for (int u = 0; u < 16; ++u) {
      aS[kf][u] = (__bf16)sv[u];
      aT[kf][u] = (__bf16)tv[u];
    }
  }

  // --- layer 1: x1 = relu(s@W1a + t@W1b + b1), two preloaded 4-chains ---
  const __bf16* W1a = wq;
  const __bf16* W1b = wq + 16384;
  const int nl = lane & 15;
#pragma unroll
  for (int nt = 0; nt < 8; ++nt) {
    v8f acc = {};
    acc = chain4(aS, W1a, nt, lane, acc);
    acc = chain4(aT, W1b, nt, lane, acc);
    epilogue_relu_store(acc, b1[nt * 16 + nl], A, nt, lane);
  }

  // --- layers 2 and 3 ---
  v16bf aF[4];
  load_afrags(A, aF, lane);
  mlp_layer(aF, wq + 32768, b2, A, lane);
  load_afrags(A, aF, lane);
  mlp_layer(aF, wq + 49152, b3, A, lane);
  load_afrags(A, aF, lane);

  // --- head: [16x128] @ [128x16(pad of 6)] ---
  const __bf16* WoT = wq + 65536;
  v16bf h0 = load_bfrag(WoT, 0, lane);
  v16bf h1v = load_bfrag(WoT, 1, lane);
  v16bf h2v = load_bfrag(WoT, 2, lane);
  v16bf h3v = load_bfrag(WoT, 3, lane);
  v8f acc = {};
  acc = __builtin_amdgcn_wmma_f32_16x16x32_bf16(false, aF[0], false, h0,
                                                (short)0, acc, false, false);
  acc = __builtin_amdgcn_wmma_f32_16x16x32_bf16(false, aF[1], false, h1v,
                                                (short)0, acc, false, false);
  acc = __builtin_amdgcn_wmma_f32_16x16x32_bf16(false, aF[2], false, h2v,
                                                (short)0, acc, false, false);
  acc = __builtin_amdgcn_wmma_f32_16x16x32_bf16(false, aF[3], false, h3v,
                                                (short)0, acc, false, false);
  float bv = bo[nl < 6 ? nl : 0];
#pragma unroll
  for (int j = 0; j < 8; ++j) {
    int r = rowbase + j + kh * 8;
    if (r < N_IMP_C && nl < 6) out[r * 6 + nl] = acc[j] + bv;
  }
}

extern "C" void kernel_launch(void* const* d_in, const int* in_sizes, int n_in,
                              void* d_out, int out_size, void* d_ws, size_t ws_size,
                              hipStream_t stream) {
  const float* h   = (const float*)d_in[0];
  const int*  idx0 = (const int*)d_in[1];
  const int*  idx1 = (const int*)d_in[2];
  const int*  idx2 = (const int*)d_in[3];
  const int*  idx3 = (const int*)d_in[4];
  const float* W1  = (const float*)d_in[5];
  const float* b1  = (const float*)d_in[6];
  const float* W2  = (const float*)d_in[7];
  const float* b2  = (const float*)d_in[8];
  const float* W3  = (const float*)d_in[9];
  const float* b3  = (const float*)d_in[10];
  const float* Wo  = (const float*)d_in[11];
  const float* bo  = (const float*)d_in[12];
  __bf16* ws = (__bf16*)d_ws;          // needs ~132 KB of scratch
  float*  out = (float*)d_out;

  prep_weights<<<64, 256, 0, stream>>>(W1, W2, W3, Wo, ws);
  const int blocks = (N_IMP_C + 127) / 128;
  janossy_mlp<<<blocks, 256, 0, stream>>>(h, idx0, idx1, idx2, idx3,
                                          b1, b2, b3, bo, ws, out);
}
